// SNNHidden_63479616635537
// MI455X (gfx1250) — matrix operand order repaired and checked by reading, not compiled, not run on hardware
//
#include <hip/hip_runtime.h>
#include <hip/hip_bf16.h>
#include <stddef.h>

// ---------------------------------------------------------------------------
// SNN LIF scan, restructured for MI455X (gfx1250, wave32, WMMA):
//   Kernel P : transpose w_rec -> wrT[k][h]              (one-time, L2-resident)
//   Kernel A : xin = x @ w_in^T  via v_wmma_f32_16x16x32_bf16 (bulk FLOPs)
//   Kernel B : sequential LIF scan; recurrent term = sparse gather of wrT rows
//              selected by a wave32-ballot spike bitmask (z is binary!)
// ---------------------------------------------------------------------------

typedef __attribute__((ext_vector_type(16))) __bf16 v16bf;
typedef __attribute__((ext_vector_type(8)))  float  v8f;

#define DT_C   0.001f
#define V_TH_C 1.0f
#define Hdim   256
#define Bdim   64

union Frag { uint4 q[2]; v16bf v; };

static __device__ __forceinline__ unsigned short f2bf(float f) {
    unsigned int u = __float_as_uint(f);
    unsigned int rnd = (u >> 16) & 1u;          // round-to-nearest-even
    u += 0x7FFFu + rnd;
    return (unsigned short)(u >> 16);
}

// ---------------------------------------------------------------------------
// Kernel P: wrT[k*256 + h] = w_rec[h*256 + k]   (so spike k contributes a
// contiguous, coalesced 1KB row to all 256 output neurons). 256KB -> lives in L2.
// ---------------------------------------------------------------------------
__global__ __launch_bounds__(256) void prep_wrec(const float* __restrict__ w_rec,
                                                 float* __restrict__ wrT) {
    int idx = blockIdx.x * 256 + threadIdx.x;   // 65536 total
    int h = idx >> 8;
    int k = idx & 255;
    wrT[(k << 8) + h] = w_rec[idx];
}

// ---------------------------------------------------------------------------
// Kernel A: xin[m, n] = sum_k x[m,k] * w_in[n,k],  M = B*T = 128000 rows.
// Workgroup: 256 threads (8 wave32 waves). Per iteration handles 32 rows x 256 cols.
//   wave w: m_sub = w>>2 (16-row half), n_base = (w&3)*64 -> four 16x16 tiles.
// w_in staged once in LDS as bf16 (padded stride 264 -> conflict-free b128 frag loads).
// A block staged per iteration. 8 K-steps of 32 -> 8 WMMA per tile.
// ---------------------------------------------------------------------------
__global__ __launch_bounds__(256) void input_gemm(const float* __restrict__ x,
                                                  const float* __restrict__ w_in,
                                                  float* __restrict__ xin,
                                                  int M) {
    __shared__ unsigned short lds_w[256 * 264];   // w_in bf16, row n, padded
    __shared__ unsigned short lds_a[32 * 264];    // A block bf16, padded

    const int tid  = threadIdx.x;
    const int wave = tid >> 5;
    const int lane = tid & 31;

    // ---- stage w_in (f32 -> bf16) once ----
    for (int idx = tid; idx < 256 * 256; idx += 256) {
        int n = idx >> 8, k = idx & 255;
        lds_w[n * 264 + k] = f2bf(w_in[idx]);
    }
    __syncthreads();

    const int m_sub  = wave >> 2;          // 0..1
    const int n_base = (wave & 3) * 64;    // 0,64,128,192
    const int mblocks = M >> 5;            // M / 32

    for (int mb = blockIdx.x; mb < mblocks; mb += gridDim.x) {
        const int m0 = mb << 5;

        // ---- stage 32x256 A block (f32 -> bf16), coalesced global reads ----
        for (int idx = tid; idx < 32 * 256; idx += 256) {
            int r = idx >> 8, k = idx & 255;
            lds_a[r * 264 + k] = f2bf(x[(size_t)(m0 + r) * Hdim + k]);
        }
        __syncthreads();

        v8f acc0 = {0.f,0.f,0.f,0.f,0.f,0.f,0.f,0.f};
        v8f acc1 = acc0, acc2 = acc0, acc3 = acc0;

        const int arow = m_sub * 16 + (lane & 15);
        const int ahalf = (lane >> 4) << 4;      // 0 or 16 (K sub-run select)
        const int bhalf = (lane >> 4) << 4;      // 0 or 16 (K sub-run select)
        const int bcol  = lane & 15;

#pragma unroll
        for (int ks = 0; ks < 8; ++ks) {
            const int kb = (ks << 5) + ahalf;
            Frag a;
            a.q[0] = *(const uint4*)&lds_a[arow * 264 + kb];        // K = kb..kb+7
            a.q[1] = *(const uint4*)&lds_a[arow * 264 + kb + 8];    // K = kb+8..kb+15

            const int kc = (ks << 5) + bhalf;                       // 16 contig K per lane
            Frag b0, b1, b2, b3;
            b0.q[0] = *(const uint4*)&lds_w[(n_base +  0 + bcol) * 264 + kc];
            b0.q[1] = *(const uint4*)&lds_w[(n_base +  0 + bcol) * 264 + kc + 8];
            b1.q[0] = *(const uint4*)&lds_w[(n_base + 16 + bcol) * 264 + kc];
            b1.q[1] = *(const uint4*)&lds_w[(n_base + 16 + bcol) * 264 + kc + 8];
            b2.q[0] = *(const uint4*)&lds_w[(n_base + 32 + bcol) * 264 + kc];
            b2.q[1] = *(const uint4*)&lds_w[(n_base + 32 + bcol) * 264 + kc + 8];
            b3.q[0] = *(const uint4*)&lds_w[(n_base + 48 + bcol) * 264 + kc];
            b3.q[1] = *(const uint4*)&lds_w[(n_base + 48 + bcol) * 264 + kc + 8];

            acc0 = __builtin_amdgcn_wmma_f32_16x16x32_bf16(false, a.v, false, b0.v, (short)0, acc0, false, false);
            acc1 = __builtin_amdgcn_wmma_f32_16x16x32_bf16(false, a.v, false, b1.v, (short)0, acc1, false, false);
            acc2 = __builtin_amdgcn_wmma_f32_16x16x32_bf16(false, a.v, false, b2.v, (short)0, acc2, false, false);
            acc3 = __builtin_amdgcn_wmma_f32_16x16x32_bf16(false, a.v, false, b3.v, (short)0, acc3, false, false);
        }

        // ---- store D tiles: VGPR r -> M = r + 8*(lane>=16); col = lane&15 ----
        const int rsel = (lane >> 4) << 3;
        const int colh = lane & 15;
#pragma unroll
        for (int r = 0; r < 8; ++r) {
            size_t row = (size_t)(m0 + m_sub * 16 + r + rsel) * Hdim;
            xin[row + n_base +  0 + colh] = acc0[r];
            xin[row + n_base + 16 + colh] = acc1[r];
            xin[row + n_base + 32 + colh] = acc2[r];
            xin[row + n_base + 48 + colh] = acc3[r];
        }
        __syncthreads();   // before re-staging lds_a
    }
}

// ---------------------------------------------------------------------------
// Kernel B: sequential LIF scan. One WG per batch element (64 WGs), one thread
// per neuron h. Recurrent term uses the binary structure of z: wave32 ballots
// build a 256-bit spike mask in LDS; each thread gathers only active wrT rows
// (L2-resident, coalesced across h). zmask holds z from the PREVIOUS step,
// matching the reference carry semantics (i_new uses old z).
// ---------------------------------------------------------------------------
__global__ __launch_bounds__(256) void lif_scan(const float* __restrict__ xin,
                                                const float* __restrict__ wrT,
                                                const float* __restrict__ tau_syn,
                                                const float* __restrict__ tau_mem,
                                                float* __restrict__ out,
                                                int T) {
    __shared__ unsigned int zmask[8];

    const int b    = blockIdx.x;
    const int h    = threadIdx.x;
    const int wave = h >> 5;
    const int lane = h & 31;

    const float ts = fminf(fmaxf(tau_syn[h], 0.f), 1.f);
    const float tm = fminf(fmaxf(tau_mem[h], 0.f), 1.f);

    float v = 0.f, cur = 0.f, z = 0.f;

    if (h < 8) zmask[h] = 0u;   // z0 = 0
    __syncthreads();

    const float* xrow = xin + (size_t)b * T * Hdim + h;
    float*       orow = out + (size_t)b * Hdim + h;
    const size_t ostride = (size_t)Bdim * Hdim;

    for (int t = 0; t < T; ++t) {
        if ((t & 7) == 0)
            __builtin_prefetch(&xrow[(size_t)(t + 16) * Hdim], 0, 1);

        // LIF decay
        float v_dec = v + DT_C * tm * (cur - v);
        float i_dec = cur - DT_C * ts * cur;

        // recurrent input from PREVIOUS spikes (zmask) + precomputed x@w_in^T
        float acc = i_dec + xrow[(size_t)t * Hdim];
#pragma unroll
        for (int w8 = 0; w8 < 8; ++w8) {
            unsigned int mm = zmask[w8];          // wave-uniform
            const int kbase = w8 << 5;
            while (mm) {
                int k = __builtin_ctz(mm);
                mm &= mm - 1;
                acc += wrT[(size_t)((kbase + k) << 8) + h];
            }
        }

        // threshold + reset
        float znew = (v_dec > V_TH_C) ? 1.f : 0.f;
        v   = (znew > 0.f) ? 0.f : v_dec;
        cur = acc;

        orow[0] = znew;          // out[t, b, h]
        orow += ostride;

        // publish znew as next step's mask
        __syncthreads();         // all waves done reading old zmask
        unsigned int m = __builtin_amdgcn_ballot_w32(znew > 0.f);
        if (lane == 0) zmask[wave] = m;
        __syncthreads();

        z = znew;
    }

    // final state (z_f, v_f, i_f) appended after out[T,B,H]
    const size_t base = (size_t)T * Bdim * Hdim;
    out[base + 0 * (size_t)Bdim * Hdim + (size_t)b * Hdim + h] = z;
    out[base + 1 * (size_t)Bdim * Hdim + (size_t)b * Hdim + h] = v;
    out[base + 2 * (size_t)Bdim * Hdim + (size_t)b * Hdim + h] = cur;
}

// ---------------------------------------------------------------------------
extern "C" void kernel_launch(void* const* d_in, const int* in_sizes, int n_in,
                              void* d_out, int out_size, void* d_ws, size_t ws_size,
                              hipStream_t stream) {
    const float* x       = (const float*)d_in[0];   // [B, T, H]
    const float* w_in    = (const float*)d_in[1];   // [H, H]
    const float* w_rec   = (const float*)d_in[2];   // [H, H]
    const float* tau_syn = (const float*)d_in[3];   // [1, H]
    const float* tau_mem = (const float*)d_in[4];   // [1, H]

    const int M = in_sizes[0] / Hdim;               // B*T
    const int T = M / Bdim;

    // workspace layout: xin [M,H] f32, then wrT [H,H] f32
    float* xin = (float*)d_ws;
    float* wrT = (float*)((char*)d_ws + (size_t)M * Hdim * sizeof(float));

    prep_wrec<<<256, 256, 0, stream>>>(w_rec, wrT);
    input_gemm<<<1024, 256, 0, stream>>>(x, w_in, xin, M);
    lif_scan<<<Bdim, 256, 0, stream>>>(xin, wrT, tau_syn, tau_mem, (float*)d_out, T);
}
